// Encoder_55997783605791
// MI455X (gfx1250) — compile-verified, matrix operand order checked by reading, not run on hardware
//
#include <hip/hip_runtime.h>
#include <hip/hip_bf16.h>

#define BQ     2
#define S_LEN  2048
#define DM     1024
#define DFF    4096
#define HEADS  16
#define HD     64
#define TOK    (BQ * S_LEN)

// 1/sqrt(64) * log2(e): scores scaled into log2 domain so v_exp_f32 (2^x) is exact
#define SCORE_SCALE 0.18033688011112042f

typedef __attribute__((ext_vector_type(16))) __bf16 v16bf;
typedef __attribute__((ext_vector_type(8)))  float  v8f;

union Frag16 {
    v16bf v;
    unsigned short u[16];
    uint4 q[2];
};

__device__ __forceinline__ unsigned short f2bf(float x) {
    unsigned int u = __float_as_uint(x);
    unsigned int r = (u + 0x7FFFu + ((u >> 16) & 1u)) >> 16;   // round-to-nearest-even
    return (unsigned short)r;
}

__device__ __forceinline__ v8f zero8() {
    v8f z;
    #pragma unroll
    for (int i = 0; i < 8; ++i) z[i] = 0.0f;
    return z;
}

__device__ __forceinline__ v8f wmma_bf16(v16bf a, v16bf b, v8f c) {
    return __builtin_amdgcn_wmma_f32_16x16x32_bf16(
        /*neg_a=*/false, a, /*neg_b=*/false, b,
        /*c_mod=*/(short)0, c, /*reuse_a=*/false, /*reuse_b=*/false);
}

// ---------------------------------------------------------------------------
// fp32 -> bf16 conversion
// ---------------------------------------------------------------------------
__global__ void cvt_bf16(const float* __restrict__ src,
                         unsigned short* __restrict__ dst, int n) {
    int i = blockIdx.x * 256 + threadIdx.x;
    if (i < n) dst[i] = f2bf(src[i]);
}

// ---------------------------------------------------------------------------
// NT GEMM: out[M,N] = A[M,K](bf16,row-major) x W[N,K](bf16,row-major)^T + bias
// 256 threads = 8 waves (4x2); wave tile 32x64 (2x4 WMMA); block tile 128x128.
// K staged 64 per barrier pair (two 32-K WMMA substeps per stage).
// ---------------------------------------------------------------------------
template<bool RELU, bool OUTBF16>
__global__ void __launch_bounds__(256)
gemm_nt(const unsigned short* __restrict__ A,
        const unsigned short* __restrict__ W,
        const float* __restrict__ bias,
        float* __restrict__ outF,
        unsigned short* __restrict__ outB,
        int M, int N, int K) {
    __shared__ __align__(16) unsigned short lA[128 * 72];
    __shared__ __align__(16) unsigned short lB[128 * 72];

    const int tid = threadIdx.x;
    const int wave = tid >> 5, lane = tid & 31;
    const int lane15 = lane & 15, laneHi = lane >> 4;
    const int koff = laneHi * 8;
    const int wr = wave >> 1, wc = wave & 1;           // 4x2 wave grid
    const int bm = blockIdx.y * 128, bn = blockIdx.x * 128;

    v8f acc[2][4];
    #pragma unroll
    for (int m = 0; m < 2; ++m)
        #pragma unroll
        for (int n = 0; n < 4; ++n) acc[m][n] = zero8();

    for (int kt = 0; kt < K; kt += 64) {
        __syncthreads();
        #pragma unroll
        for (int c = 0; c < 4; ++c) {
            int e = tid * 8 + c * 2048;
            int row = e >> 6, col = e & 63;
            *(uint4*)(lA + row * 72 + col) =
                *(const uint4*)(A + (size_t)(bm + row) * K + kt + col);
            *(uint4*)(lB + row * 72 + col) =
                *(const uint4*)(W + (size_t)(bn + row) * K + kt + col);
        }
        __syncthreads();

        #pragma unroll
        for (int kk = 0; kk < 2; ++kk) {
            Frag16 af[2], bf2[4];
            #pragma unroll
            for (int m = 0; m < 2; ++m) {
                const unsigned short* p =
                    lA + (wr * 32 + m * 16 + lane15) * 72 + kk * 32;
                af[m].q[0] = *(const uint4*)(p + koff);
                af[m].q[1] = *(const uint4*)(p + koff + 16);
            }
            #pragma unroll
            for (int n = 0; n < 4; ++n) {
                const unsigned short* p =
                    lB + (wc * 64 + n * 16 + lane15) * 72 + kk * 32;
                bf2[n].q[0] = *(const uint4*)(p + koff);
                bf2[n].q[1] = *(const uint4*)(p + koff + 16);
            }
            #pragma unroll
            for (int m = 0; m < 2; ++m)
                #pragma unroll
                for (int n = 0; n < 4; ++n)
                    acc[m][n] = wmma_bf16(af[m].v, bf2[n].v, acc[m][n]);
        }
    }

    #pragma unroll
    for (int m = 0; m < 2; ++m)
        #pragma unroll
        for (int n = 0; n < 4; ++n)
            #pragma unroll
            for (int r = 0; r < 8; ++r) {
                int row = bm + wr * 32 + m * 16 + r + laneHi * 8;
                int col = bn + wc * 64 + n * 16 + lane15;
                float v = acc[m][n][r] + bias[col];
                if (RELU) v = fmaxf(v, 0.0f);
                if (OUTBF16) outB[(size_t)row * N + col] = f2bf(v);
                else         outF[(size_t)row * N + col] = v;
            }
}

// ---------------------------------------------------------------------------
// V transpose: vT[b*S+s][h*64+d] -> vtt[((b*H+h)*64+d)*S + s]  (bf16)
// ---------------------------------------------------------------------------
__global__ void __launch_bounds__(256)
transpose_v(const unsigned short* __restrict__ vT,
            unsigned short* __restrict__ vtt) {
    __shared__ __align__(16) unsigned short t[64 * 72];
    const int sblk = blockIdx.x * 64;
    const int h = blockIdx.y, b = blockIdx.z;
    const int tid = threadIdx.x;

    #pragma unroll
    for (int c = 0; c < 2; ++c) {
        int e = tid * 8 + c * 2048;
        int sr = e >> 6, dc = e & 63;
        *(uint4*)(t + sr * 72 + dc) =
            *(const uint4*)(vT + (size_t)(b * S_LEN + sblk + sr) * DM + h * HD + dc);
    }
    __syncthreads();
    #pragma unroll
    for (int c = 0; c < 2; ++c) {
        int e = tid * 8 + c * 2048;
        int d = e >> 6, s0 = e & 63;
        uint4 val;
        unsigned short* pv = (unsigned short*)&val;
        #pragma unroll
        for (int j = 0; j < 8; ++j) pv[j] = t[(s0 + j) * 72 + d];
        *(uint4*)(vtt + ((size_t)(b * HEADS + h) * HD + d) * S_LEN + sblk + s0) = val;
    }
}

// ---------------------------------------------------------------------------
// Flash attention: block = (qblock of 64 rows, head, batch), 128 thr = 4 waves.
// Each wave: 16 q rows, online softmax over S in 64-key steps (4 score tiles),
// amortizing reductions/rescale over 2x keys. exp in log2 domain (v_exp_f32).
// K / V^T fragments come straight from global (L2-resident rows).
// ---------------------------------------------------------------------------
__global__ void __launch_bounds__(128)
attn_kernel(const unsigned short* __restrict__ qT,
            const unsigned short* __restrict__ kT,
            const unsigned short* __restrict__ vtt,
            const int* __restrict__ mask,
            unsigned short* __restrict__ ctxb) {
    __shared__ __align__(16) unsigned short lP[4 * 16 * 64];   // per-wave P scratch

    const int tid = threadIdx.x;
    const int wave = tid >> 5, lane = tid & 31;
    const int lane15 = lane & 15, laneHi = lane >> 4;
    const int koff = laneHi * 8;
    const int qblk = blockIdx.x, h = blockIdx.y, b = blockIdx.z;
    const int qrow0 = qblk * 64 + wave * 16;

    // Q fragments for this wave's 16 rows (16x64 = two 16x32 A-frags)
    Frag16 aq[2];
    {
        const unsigned short* qp =
            qT + (size_t)(b * S_LEN + qrow0 + lane15) * DM + h * HD;
        #pragma unroll
        for (int kk = 0; kk < 2; ++kk) {
            aq[kk].q[0] = *(const uint4*)(qp + kk * 32 + koff);
            aq[kk].q[1] = *(const uint4*)(qp + kk * 32 + koff + 16);
        }
    }

    float mst[8], lst[8];
    v8f acc[4];
    #pragma unroll
    for (int r = 0; r < 8; ++r) { mst[r] = -1e30f; lst[r] = 0.0f; }
    #pragma unroll
    for (int n = 0; n < 4; ++n) acc[n] = zero8();

    const unsigned short* kbase = kT + (size_t)(b * S_LEN) * DM + h * HD;
    const unsigned short* vtbase = vtt + (size_t)(b * HEADS + h) * HD * S_LEN;
    unsigned short* myP = lP + wave * 16 * 64;
    const int* mrow = mask + b * S_LEN;

    for (int kb = 0; kb < S_LEN; kb += 64) {
        // scores: four 16x16 tiles covering 64 keys
        v8f sc[4];
        int mk[4];
        #pragma unroll
        for (int nh = 0; nh < 4; ++nh) {
            sc[nh] = zero8();
            const unsigned short* kp =
                kbase + (size_t)(kb + nh * 16 + lane15) * DM;
            #pragma unroll
            for (int dk = 0; dk < 2; ++dk) {
                Frag16 bk;
                bk.q[0] = *(const uint4*)(kp + dk * 32 + koff);
                bk.q[1] = *(const uint4*)(kp + dk * 32 + koff + 16);
                sc[nh] = wmma_bf16(aq[dk].v, bk.v, sc[nh]);
            }
            mk[nh] = mrow[kb + nh * 16 + lane15];
        }

        // online softmax update over 64 keys (log2 domain)
        #pragma unroll
        for (int r = 0; r < 8; ++r) {
            float s[4];
            #pragma unroll
            for (int nh = 0; nh < 4; ++nh) {
                s[nh] = sc[nh][r] * SCORE_SCALE;
                if (mk[nh] == 0) s[nh] = -1e9f;
            }
            float rm = fmaxf(fmaxf(s[0], s[1]), fmaxf(s[2], s[3]));
            #pragma unroll
            for (int m = 1; m < 16; m <<= 1) rm = fmaxf(rm, __shfl_xor(rm, m, 32));
            float mnew = fmaxf(mst[r], rm);
            float sc_old = exp2f(mst[r] - mnew);
            float e[4];
            float rs = 0.0f;
            #pragma unroll
            for (int nh = 0; nh < 4; ++nh) {
                e[nh] = exp2f(s[nh] - mnew);
                rs += e[nh];
            }
            #pragma unroll
            for (int m = 1; m < 16; m <<= 1) rs += __shfl_xor(rs, m, 32);
            lst[r] = lst[r] * sc_old + rs;
            mst[r] = mnew;
            #pragma unroll
            for (int n = 0; n < 4; ++n) acc[n][r] *= sc_old;
            // stash P row into wave-private scratch (bf16, A-frag source layout)
            int prow = r + laneHi * 8;
            #pragma unroll
            for (int nh = 0; nh < 4; ++nh)
                myP[prow * 64 + nh * 16 + lane15] = f2bf(e[nh]);
        }

        // ctx += P(16x64) x V(64x64): two K-substeps of 32 keys each
        #pragma unroll
        for (int kk = 0; kk < 2; ++kk) {
            Frag16 ap;
            const unsigned short* pp = myP + lane15 * 64 + kk * 32;
            ap.q[0] = *(const uint4*)(pp + koff);
            ap.q[1] = *(const uint4*)(pp + koff + 16);
            #pragma unroll
            for (int nc = 0; nc < 4; ++nc) {
                Frag16 fv;
                const unsigned short* vp =
                    vtbase + (size_t)(nc * 16 + lane15) * S_LEN + kb + kk * 32;
                fv.q[0] = *(const uint4*)(vp + koff);
                fv.q[1] = *(const uint4*)(vp + koff + 16);
                acc[nc] = wmma_bf16(ap.v, fv.v, acc[nc]);
            }
        }
    }

    // normalize and store ctx as bf16 ([B,S,D] layout)
    #pragma unroll
    for (int nc = 0; nc < 4; ++nc)
        #pragma unroll
        for (int r = 0; r < 8; ++r) {
            int tok = b * S_LEN + qrow0 + r + laneHi * 8;
            int d = h * HD + nc * 16 + lane15;
            ctxb[(size_t)tok * DM + d] = f2bf(acc[nc][r] / lst[r]);
        }
}

// ---------------------------------------------------------------------------
// fused residual-add + LayerNorm (fp32 out, optional bf16 copy)
// ---------------------------------------------------------------------------
__global__ void __launch_bounds__(256)
add_ln(const float* __restrict__ a, const float* __restrict__ res,
       const float* __restrict__ g, const float* __restrict__ beta,
       float* __restrict__ outF, unsigned short* __restrict__ outB) {
    __shared__ float red[256];
    const int row = blockIdx.x;
    const int tid = threadIdx.x;
    const float* pa = a + (size_t)row * DM;
    const float* pr = res + (size_t)row * DM;

    float v[4];
    float s = 0.0f;
    #pragma unroll
    for (int i = 0; i < 4; ++i) {
        int c = tid + i * 256;
        v[i] = pa[c] + pr[c];
        s += v[i];
    }
    red[tid] = s;
    __syncthreads();
    for (int st = 128; st > 0; st >>= 1) {
        if (tid < st) red[tid] += red[tid + st];
        __syncthreads();
    }
    const float mu = red[0] * (1.0f / DM);
    __syncthreads();

    float sq = 0.0f;
    #pragma unroll
    for (int i = 0; i < 4; ++i) {
        float d = v[i] - mu;
        sq += d * d;
    }
    red[tid] = sq;
    __syncthreads();
    for (int st = 128; st > 0; st >>= 1) {
        if (tid < st) red[tid] += red[tid + st];
        __syncthreads();
    }
    const float var = red[0] * (1.0f / DM);
    const float rs = rsqrtf(var + 1e-5f);

    #pragma unroll
    for (int i = 0; i < 4; ++i) {
        int c = tid + i * 256;
        float y = (v[i] - mu) * rs * g[c] + beta[c];
        outF[(size_t)row * DM + c] = y;
        if (outB) outB[(size_t)row * DM + c] = f2bf(y);
    }
}

// ---------------------------------------------------------------------------
extern "C" void kernel_launch(void* const* d_in, const int* in_sizes, int n_in,
                              void* d_out, int out_size, void* d_ws, size_t ws_size,
                              hipStream_t stream) {
    (void)in_sizes; (void)n_in; (void)out_size; (void)ws_size;
    const float* x    = (const float*)d_in[0];
    const int*   mask = (const int*)d_in[1];
    const float* Wq = (const float*)d_in[2];  const float* bq = (const float*)d_in[3];
    const float* Wk = (const float*)d_in[4];  const float* bk = (const float*)d_in[5];
    const float* Wv = (const float*)d_in[6];  const float* bv = (const float*)d_in[7];
    const float* Wo = (const float*)d_in[8];  const float* bo = (const float*)d_in[9];
    const float* W1 = (const float*)d_in[10]; const float* b1 = (const float*)d_in[11];
    const float* W2 = (const float*)d_in[12]; const float* b2 = (const float*)d_in[13];
    const float* lng = (const float*)d_in[14]; const float* lnb = (const float*)d_in[15];

    char* ws = (char*)d_ws;
    size_t off = 0;
    auto alloc = [&](size_t bytes) -> void* {
        void* p = ws + off;
        off += (bytes + 255) & ~(size_t)255;
        return p;
    };

    unsigned short* xb  = (unsigned short*)alloc((size_t)TOK * DM * 2);
    unsigned short* wqb = (unsigned short*)alloc((size_t)DM * DM * 2);
    unsigned short* wkb = (unsigned short*)alloc((size_t)DM * DM * 2);
    unsigned short* wvb = (unsigned short*)alloc((size_t)DM * DM * 2);
    unsigned short* wob = (unsigned short*)alloc((size_t)DM * DM * 2);
    unsigned short* w1b = (unsigned short*)alloc((size_t)DM * DFF * 2);
    unsigned short* w2b = (unsigned short*)alloc((size_t)DM * DFF * 2);
    unsigned short* qT  = (unsigned short*)alloc((size_t)TOK * DM * 2);
    unsigned short* kT  = (unsigned short*)alloc((size_t)TOK * DM * 2);
    unsigned short* vT  = (unsigned short*)alloc((size_t)TOK * DM * 2);
    unsigned short* vtt = (unsigned short*)alloc((size_t)TOK * DM * 2);
    unsigned short* ctxb = (unsigned short*)alloc((size_t)TOK * DM * 2);
    float*          hbuf = (float*)alloc((size_t)TOK * DM * 4);
    unsigned short* hb   = (unsigned short*)alloc((size_t)TOK * DM * 2);
    unsigned short* ffb  = (unsigned short*)alloc((size_t)TOK * DFF * 2);
    // fp32 residual scratch reuses the (dead-by-then) q/k bf16 region:
    // qT..kT spans 2 * TOK*DM*2 bytes = TOK*DM*4 bytes, exactly one fp32 buffer.
    float* attn_out = (float*)qT;
    float* ff2      = (float*)qT;

    // ---- 1) fp32 -> bf16 conversions
    cvt_bf16<<<(TOK * DM + 255) / 256, 256, 0, stream>>>(x,  xb,  TOK * DM);
    cvt_bf16<<<(DM * DM + 255) / 256, 256, 0, stream>>>(Wq, wqb, DM * DM);
    cvt_bf16<<<(DM * DM + 255) / 256, 256, 0, stream>>>(Wk, wkb, DM * DM);
    cvt_bf16<<<(DM * DM + 255) / 256, 256, 0, stream>>>(Wv, wvb, DM * DM);
    cvt_bf16<<<(DM * DM + 255) / 256, 256, 0, stream>>>(Wo, wob, DM * DM);
    cvt_bf16<<<(DM * DFF + 255) / 256, 256, 0, stream>>>(W1, w1b, DM * DFF);
    cvt_bf16<<<(DM * DFF + 255) / 256, 256, 0, stream>>>(W2, w2b, DM * DFF);

    // ---- 2) QKV projections (bf16 out)
    dim3 g1(DM / 128, TOK / 128);
    gemm_nt<false, true><<<g1, 256, 0, stream>>>(xb, wqb, bq, nullptr, qT, TOK, DM, DM);
    gemm_nt<false, true><<<g1, 256, 0, stream>>>(xb, wkb, bk, nullptr, kT, TOK, DM, DM);
    gemm_nt<false, true><<<g1, 256, 0, stream>>>(xb, wvb, bv, nullptr, vT, TOK, DM, DM);

    // ---- 3) transpose V for contiguous key-major fragments
    dim3 gt(S_LEN / 64, HEADS, BQ);
    transpose_v<<<gt, 256, 0, stream>>>(vT, vtt);

    // ---- 4) attention
    dim3 ga(S_LEN / 64, HEADS, BQ);
    attn_kernel<<<ga, 128, 0, stream>>>(qT, kT, vtt, mask, ctxb);

    // ---- 5) output projection (fp32 out, into reused region)
    gemm_nt<false, false><<<g1, 256, 0, stream>>>(ctxb, wob, bo, attn_out, nullptr, TOK, DM, DM);

    // ---- 6) h = LN(attn_out + x), fp32 + bf16 copies
    add_ln<<<TOK, 256, 0, stream>>>(attn_out, x, lng, lnb, hbuf, hb);

    // ---- 7) FFN1 with ReLU (bf16 out)
    dim3 g2(DFF / 128, TOK / 128);
    gemm_nt<true, true><<<g2, 256, 0, stream>>>(hb, w1b, b1, nullptr, ffb, TOK, DFF, DM);

    // ---- 8) FFN2 (fp32 out, reuse region; attn_out is dead after step 6)
    gemm_nt<false, false><<<g1, 256, 0, stream>>>(ffb, w2b, b2, ff2, nullptr, TOK, DM, DFF);

    // ---- 9) out = LN(ff2 + h)
    add_ln<<<TOK, 256, 0, stream>>>(ff2, hbuf, lng, lnb, (float*)d_out, nullptr);
}
